// TheoLIF_Executor_84945863180429
// MI455X (gfx1250) — matrix operand order repaired and checked by reading, not compiled, not run on hardware
//
#include <hip/hip_runtime.h>

// ---------------------------------------------------------------------------
// LIF spiking network on MI455X (gfx1250, wave32, WMMA).
// B=8192, H=1024.  drive = X @ Wi once; 25 steps of v = .7v + .3(s@Wr) + drive.
// bf16 split-precision WMMA (W = Whi + Wlo) => ~f32 accuracy, bf16 throughput.
// Hot step kernel stages 128x64 weight tiles in LDS (double-buffered, one
// barrier per 32 WMMAs) to cut L2 traffic 4x and keep the WMMA pipes fed.
// ---------------------------------------------------------------------------

typedef __bf16 bf16_t;
typedef __attribute__((ext_vector_type(16))) __bf16        v16bf;
typedef __attribute__((ext_vector_type(8)))  float         v8f;
typedef __attribute__((ext_vector_type(4)))  unsigned int  u32x4;

union Frag { v16bf v; u32x4 q[2]; };

#define H_DIM 1024
#define B_DIM 8192
#define BETA 0.7f
#define ONE_MINUS_BETA 0.3f
#define THRESHOLD 1.0f
#define KC 64            // K elements staged per phase (2 WMMA sub-chunks)
#define LDS_STRIDE 72    // 64 bf16 + 8 pad => bank-conflict-free b128 frag reads

static __device__ __forceinline__ v8f wmma_bf16(const Frag& a, const Frag& b, v8f c) {
    return __builtin_amdgcn_wmma_f32_16x16x32_bf16(
        /*neg_a=*/false, a.v, /*neg_b=*/false, b.v,
        /*c_mod=*/(short)0, c, /*reuse_a=*/false, /*reuse_b=*/false);
}

// ---- transpose + hi/lo bf16 split of a weight matrix W[k][n] -> Wt[n][k] ----
__global__ __launch_bounds__(256) void prep_weight_t(const float* __restrict__ W,
                                                     bf16_t* __restrict__ Thi,
                                                     bf16_t* __restrict__ Tlo) {
    int i = blockIdx.x * 256 + threadIdx.x;      // over H*H, layout n*H + k
    int k = i & (H_DIM - 1);
    int n = i >> 10;
    float w = W[(size_t)k * H_DIM + n];
    bf16_t hi = (bf16_t)w;
    bf16_t lo = (bf16_t)(w - (float)hi);
    Thi[i] = hi;
    Tlo[i] = lo;
}

// ---- split f32 state into bf16 hi/lo (row-major, A-operand source) ----------
__global__ __launch_bounds__(256) void split_state(const float* __restrict__ X,
                                                   bf16_t* __restrict__ Xhi,
                                                   bf16_t* __restrict__ Xlo) {
    size_t i = (size_t)blockIdx.x * 256 + threadIdx.x;
    float x = X[i];
    bf16_t hi = (bf16_t)x;
    Xhi[i] = hi;
    Xlo[i] = (bf16_t)(x - (float)hi);
}

// ---- step 1 is elementwise: v=drive, spike, reset ---------------------------
__global__ __launch_bounds__(256) void first_step(const float* __restrict__ drive,
                                                  float* __restrict__ v,
                                                  bf16_t* __restrict__ spikes) {
    size_t i = (size_t)blockIdx.x * 256 + threadIdx.x;
    float vn = drive[i];
    float sp = (vn >= THRESHOLD) ? 1.0f : 0.0f;
    v[i] = vn * (1.0f - sp);
    spikes[i] = (bf16_t)sp;
}

// ---- fragment loaders -------------------------------------------------------
// A (16x32 bf16, row-major source): lane ln holds row m0+ln; half=0 -> K {0..15},
// half=1 -> K {16..31}; 32 contiguous bytes.
static __device__ __forceinline__ Frag load_a(const bf16_t* __restrict__ A,
                                              int row, int k0, int half) {
    Frag f;
    const bf16_t* p = A + (size_t)row * H_DIM + k0 + half * 16;
    f.q[0] = *(const u32x4*)(p);
    f.q[1] = *(const u32x4*)(p + 8);
    return f;
}
// B (32x16 bf16) from transposed weights Wt[n][k]: lane ln -> N=n; half selects
// K chunk k0+half*16 .. +15 => 32 contiguous bytes.
static __device__ __forceinline__ Frag load_b(const bf16_t* __restrict__ Wt,
                                              int n, int k0, int half) {
    Frag f;
    const bf16_t* p = Wt + (size_t)n * H_DIM + k0 + half * 16;
    f.q[0] = *(const u32x4*)(p);
    f.q[1] = *(const u32x4*)(p + 8);
    return f;
}
// B fragment from an LDS tile (128 rows x KC k, padded stride LDS_STRIDE).
static __device__ __forceinline__ Frag load_b_lds(const bf16_t* base,
                                                  int n_local, int kc, int half) {
    Frag f;
    const bf16_t* p = base + n_local * LDS_STRIDE + kc * 32 + half * 16;
    f.q[0] = *(const u32x4*)(p);
    f.q[1] = *(const u32x4*)(p + 8);
    return f;
}
// Stage a 128xKC bf16 weight tile into LDS: thread t covers row t>>1,
// k-range (t&1)*32 .. +31 (four b128 global loads + four b128 LDS stores).
static __device__ __forceinline__ void stage_tile(const bf16_t* __restrict__ Wt,
                                                  bf16_t* dst, int n_base, int k0,
                                                  int tid) {
    const int n_local = tid >> 1;
    const int ks = (tid & 1) * 32;
    const bf16_t* src = Wt + (size_t)(n_base + n_local) * H_DIM + k0 + ks;
    u32x4 d0 = *(const u32x4*)(src);
    u32x4 d1 = *(const u32x4*)(src + 8);
    u32x4 d2 = *(const u32x4*)(src + 16);
    u32x4 d3 = *(const u32x4*)(src + 24);
    bf16_t* d = dst + n_local * LDS_STRIDE + ks;
    *(u32x4*)(d)      = d0;
    *(u32x4*)(d + 8)  = d1;
    *(u32x4*)(d + 16) = d2;
    *(u32x4*)(d + 24) = d3;
}

// ---- drive GEMM: drive = Ahi@(Whi+Wlo) + Alo@Whi  (3-term bf16 split) ------
__global__ __launch_bounds__(256) void drive_gemm(const bf16_t* __restrict__ Ahi,
                                                  const bf16_t* __restrict__ Alo,
                                                  const bf16_t* __restrict__ Whi,
                                                  const bf16_t* __restrict__ Wlo,
                                                  float* __restrict__ drive) {
    const int lane = threadIdx.x & 31;
    const int wave = threadIdx.x >> 5;
    const int wm = wave >> 1, wn = wave & 1;
    const int half = lane >> 4, ln = lane & 15;
    const int m0 = blockIdx.x * 128 + wm * 32;
    const int n0 = blockIdx.y * 128 + wn * 64;

    v8f acc[2][4];
    const v8f vzero = {0.f, 0.f, 0.f, 0.f, 0.f, 0.f, 0.f, 0.f};
#pragma unroll
    for (int mf = 0; mf < 2; ++mf)
#pragma unroll
        for (int nf = 0; nf < 4; ++nf) acc[mf][nf] = vzero;

    for (int kt = 0; kt < H_DIM / 32; ++kt) {
        const int k0 = kt * 32;
        Frag ah[2], al[2], bh[4], bl[4];
#pragma unroll
        for (int mf = 0; mf < 2; ++mf) {
            ah[mf] = load_a(Ahi, m0 + mf * 16 + ln, k0, half);
            al[mf] = load_a(Alo, m0 + mf * 16 + ln, k0, half);
        }
#pragma unroll
        for (int nf = 0; nf < 4; ++nf) {
            bh[nf] = load_b(Whi, n0 + nf * 16 + ln, k0, half);
            bl[nf] = load_b(Wlo, n0 + nf * 16 + ln, k0, half);
        }
#pragma unroll
        for (int mf = 0; mf < 2; ++mf)
#pragma unroll
            for (int nf = 0; nf < 4; ++nf) {
                acc[mf][nf] = wmma_bf16(ah[mf], bh[nf], acc[mf][nf]);
                acc[mf][nf] = wmma_bf16(ah[mf], bl[nf], acc[mf][nf]);
                acc[mf][nf] = wmma_bf16(al[mf], bh[nf], acc[mf][nf]);
            }
    }
#pragma unroll
    for (int mf = 0; mf < 2; ++mf)
#pragma unroll
        for (int nf = 0; nf < 4; ++nf) {
            const int col = n0 + nf * 16 + ln;
#pragma unroll
            for (int r = 0; r < 8; ++r) {
                const int row = m0 + mf * 16 + half * 8 + r;
                drive[(size_t)row * H_DIM + col] = acc[mf][nf][r];
            }
        }
}

// ---- one LIF step: C = spikes@(Whi+Wlo); v' = b*v + .3C + drive; fire/reset -
// 128x64 weight tiles double-buffered through LDS (72 KB); one barrier per
// 32 WMMAs; staging of phase p+1 overlaps compute on phase p.
__global__ __launch_bounds__(256) void lif_step(const bf16_t* __restrict__ spikes_in,
                                                const bf16_t* __restrict__ Whi,
                                                const bf16_t* __restrict__ Wlo,
                                                const float* __restrict__ drive,
                                                float* __restrict__ v,
                                                bf16_t* __restrict__ spikes_out,
                                                float* __restrict__ final_out) {
    __shared__ __align__(16) bf16_t shB[2][2][128 * LDS_STRIDE];  // [buf][hi/lo]

    const int tid = threadIdx.x;
    const int lane = tid & 31;
    const int wave = tid >> 5;
    const int wm = wave >> 1, wn = wave & 1;
    const int half = lane >> 4, ln = lane & 15;
    const int m0 = blockIdx.x * 128 + wm * 32;
    const int nb = blockIdx.y * 128;
    const int NPHASE = H_DIM / KC;     // 16

    v8f acc[2][4];
    const v8f vzero = {0.f, 0.f, 0.f, 0.f, 0.f, 0.f, 0.f, 0.f};
#pragma unroll
    for (int mf = 0; mf < 2; ++mf)
#pragma unroll
        for (int nf = 0; nf < 4; ++nf) acc[mf][nf] = vzero;

    // prologue: stage phase 0
    stage_tile(Whi, shB[0][0], nb, 0, tid);
    stage_tile(Wlo, shB[0][1], nb, 0, tid);

    for (int kp = 0; kp < NPHASE; ++kp) {
        __syncthreads();                       // phase kp staged & prior reads done
        const int cur = kp & 1;
        if (kp + 1 < NPHASE) {                 // overlap: stage phase kp+1
            stage_tile(Whi, shB[cur ^ 1][0], nb, (kp + 1) * KC, tid);
            stage_tile(Wlo, shB[cur ^ 1][1], nb, (kp + 1) * KC, tid);
        }
#pragma unroll
        for (int kc = 0; kc < KC / 32; ++kc) {
            const int k0 = kp * KC + kc * 32;
            Frag a[2], bh[4], bl[4];
#pragma unroll
            for (int mf = 0; mf < 2; ++mf)
                a[mf] = load_a(spikes_in, m0 + mf * 16 + ln, k0, half);
#pragma unroll
            for (int nf = 0; nf < 4; ++nf) {
                const int n_local = wn * 64 + nf * 16 + ln;
                bh[nf] = load_b_lds(shB[cur][0], n_local, kc, half);
                bl[nf] = load_b_lds(shB[cur][1], n_local, kc, half);
            }
#pragma unroll
            for (int mf = 0; mf < 2; ++mf)
#pragma unroll
                for (int nf = 0; nf < 4; ++nf) {
                    acc[mf][nf] = wmma_bf16(a[mf], bh[nf], acc[mf][nf]);
                    acc[mf][nf] = wmma_bf16(a[mf], bl[nf], acc[mf][nf]);
                }
        }
    }

#pragma unroll
    for (int mf = 0; mf < 2; ++mf)
#pragma unroll
        for (int nf = 0; nf < 4; ++nf) {
            const int col = nb + wn * 64 + nf * 16 + ln;
#pragma unroll
            for (int r = 0; r < 8; ++r) {
                const int row = m0 + mf * 16 + half * 8 + r;
                const size_t idx = (size_t)row * H_DIM + col;
                float vn = BETA * v[idx] + ONE_MINUS_BETA * acc[mf][nf][r] + drive[idx];
                float sp = (vn >= THRESHOLD) ? 1.0f : 0.0f;
                v[idx] = vn * (1.0f - sp);
                spikes_out[idx] = (bf16_t)sp;
                if (final_out) final_out[idx] = sp;
            }
        }
}

// ---------------------------------------------------------------------------
extern "C" void kernel_launch(void* const* d_in, const int* in_sizes, int n_in,
                              void* d_out, int out_size, void* d_ws, size_t ws_size,
                              hipStream_t stream) {
    const float* init_state = (const float*)d_in[0];   // [8192,1024] f32
    const float* rec_w      = (const float*)d_in[1];   // [1024,1024] f32
    const float* in_w       = (const float*)d_in[2];   // [1024,1024] f32
    float* out = (float*)d_out;

    const size_t BH = (size_t)B_DIM * H_DIM;           // 8M
    const size_t HH = (size_t)H_DIM * H_DIM;           // 1M
    char* ws = (char*)d_ws;

    float*  drive = (float*)(ws);                      // 32 MB
    float*  v     = (float*)(ws + BH * 4);             // 32 MB
    bf16_t* bufA  = (bf16_t*)(ws + BH * 8);            // 16 MB (Ahi, then spikes ping)
    bf16_t* bufB  = (bf16_t*)(ws + BH * 8 + BH * 2);   // 16 MB (Alo, then spikes pong)
    bf16_t* WhiR  = (bf16_t*)(ws + BH * 12);           // 2 MB each below
    bf16_t* WloR  = WhiR + HH;
    bf16_t* WhiI  = WloR + HH;
    bf16_t* WloI  = WhiI + HH;

    // weight transpose + split (one-time, 4 MB each, trivially cheap)
    prep_weight_t<<<(int)(HH / 256), 256, 0, stream>>>(rec_w, WhiR, WloR);
    prep_weight_t<<<(int)(HH / 256), 256, 0, stream>>>(in_w, WhiI, WloI);

    // state split, drive GEMM (3-term bf16 split), elementwise step 1
    split_state<<<(int)(BH / 256), 256, 0, stream>>>(init_state, bufA, bufB);
    dim3 g(B_DIM / 128, H_DIM / 128);
    drive_gemm<<<g, 256, 0, stream>>>(bufA, bufB, WhiI, WloI, drive);
    first_step<<<(int)(BH / 256), 256, 0, stream>>>(drive, v, bufA);

    // steps 2..25 (24 GEMM steps), ping-pong spike buffers
    bf16_t* sin = bufA;
    bf16_t* sout = bufB;
    for (int s = 0; s < 24; ++s) {
        lif_step<<<g, 256, 0, stream>>>(sin, WhiR, WloR, drive, v, sout,
                                        (s == 23) ? out : nullptr);
        bf16_t* t = sin; sin = sout; sout = t;
    }
}
